// EdgeMaskGenerator_11699490914484
// MI455X (gfx1250) — matrix operand order repaired and checked by reading, not compiled, not run on hardware
//
#include <hip/hip_runtime.h>
#include <hip/hip_bf16.h>
#include <stdint.h>

// ---------- WMMA / load types ----------
typedef __bf16 bf16_t;
typedef __attribute__((ext_vector_type(16))) __bf16 v16bf;
typedef __attribute__((ext_vector_type(8)))  float  v8f;
typedef unsigned int v4u __attribute__((ext_vector_type(4)));   // raw 128-bit
typedef __attribute__((address_space(1))) const v4u gc_v4u;     // global AS load

union BfPack { v4u q[2]; v16bf v; };

__device__ __forceinline__ unsigned int f2bf1(float f) {
  unsigned int u = __float_as_uint(f);
  u += 0x7FFFu + ((u >> 16) & 1u);   // round-to-nearest-even to bf16
  return u >> 16;
}

// ---------- prep: f32 -> bf16 (8 elements / thread) ----------
__global__ void cvt_bf16_kernel(const float* __restrict__ src,
                                uint4* __restrict__ dst, int n8) {
  int i = blockIdx.x * blockDim.x + threadIdx.x;
  if (i >= n8) return;
  const float4* s4 = (const float4*)src;
  float4 a = s4[2 * i];
  float4 b = s4[2 * i + 1];
  uint4 o;
  o.x = f2bf1(a.x) | (f2bf1(a.y) << 16);
  o.y = f2bf1(a.z) | (f2bf1(a.w) << 16);
  o.z = f2bf1(b.x) | (f2bf1(b.y) << 16);
  o.w = f2bf1(b.z) | (f2bf1(b.w) << 16);
  dst[i] = o;
}

// ---------- prep: pack W1' (rows 0..255 = W1[0:256], rows 256..383 = W1[384:512])
// B-operand block (t = n-tile 0..7, s = k-step 0..11): 32 lanes x 16 bf16.
// lane l: N = t*16 + (l&15); K = s*32 + (l>>4)*16 + j  (j = 0..15, consecutive bf16)
__global__ void pack_w1_kernel(const float* __restrict__ W1,
                               unsigned short* __restrict__ W1p, int nTotal) {
  int tid = blockIdx.x * blockDim.x + threadIdx.x;
  if (tid >= nTotal) return;                 // nTotal = 8*12*32*16 = 49152
  int j = tid & 15;
  int l = (tid >> 4) & 31;
  int s = (tid >> 9) % 12;
  int t = tid / 6144;
  int kp = s * 32 + (l >> 4) * 16 + j;       // 0..383 in W1'
  int n  = t * 16 + (l & 15);
  int row = (kp < 256) ? kp : kp + 128;      // skip the h_rq block
  W1p[tid] = (unsigned short)f2bf1(W1[row * 128 + n]);
}

// ---------- prep: cvec[n] = b1[n] + sum_k rq[k] * W1[256+k][n] ----------
__global__ void calc_cvec_kernel(const float* __restrict__ rel_table,
                                 const float* __restrict__ W1,
                                 const float* __restrict__ b1,
                                 const int* __restrict__ qrel,
                                 float* __restrict__ cvec) {
  int n = threadIdx.x;                       // 128 threads
  int q = qrel[0];
  const float* rq = rel_table + (size_t)q * 128;
  float s = b1[n];
  for (int k = 0; k < 128; ++k) s += rq[k] * W1[(256 + k) * 128 + n];
  cvec[n] = s;
}

// ---------- main: 32 edges/wave, K=384 (3 segments x 4 k-steps), full N=128 ----------
__global__ __launch_bounds__(256) void edge_score_kernel(
    float* __restrict__ out,
    const int* __restrict__ srcIdx,
    const int* __restrict__ relIdx,
    const int* __restrict__ timIdx,
    const unsigned char* __restrict__ hB,
    const unsigned char* __restrict__ relB,
    const unsigned char* __restrict__ timB,
    const v4u* __restrict__ W1p,
    const float* __restrict__ cvec,
    const float* __restrict__ W2,
    const float* __restrict__ b2,
    int E) {
  extern __shared__ v4u ldsW[];              // 6144 x 16B = 96 KB packed W1'
  for (int i = threadIdx.x; i < 6144; i += 256) ldsW[i] = W1p[i];
  __syncthreads();

  const int lane = threadIdx.x & 31;
  const int wave = threadIdx.x >> 5;
  const int half = lane >> 4;
  const int lm   = lane & 15;

  const float bias2 = b2[0];

  // per-lane epilogue constants for column n = t*16 + lm (tile-invariant)
  float cv[8], w2v[8];
#pragma unroll
  for (int t = 0; t < 8; ++t) {
    cv[t]  = cvec[t * 16 + lm];
    w2v[t] = W2[t * 16 + lm];
  }

  v8f zero8;
#pragma unroll
  for (int i = 0; i < 8; ++i) zero8[i] = 0.0f;

  const int tiles = (E + 31) >> 5;
#pragma unroll 1
  for (int tile = blockIdx.x * 8 + wave; tile < tiles; tile += gridDim.x * 8) {
    const int ebase = tile * 32;

    // named row pointers: two 16-edge A tiles x 3 segments (no arrays -> no movrels)
    int e0 = ebase + lm;        if (e0 >= E) e0 = E - 1;
    int e1 = ebase + 16 + lm;   if (e1 >= E) e1 = E - 1;
    const unsigned char* pH0 = hB   + (size_t)srcIdx[e0] * 256u;
    const unsigned char* pR0 = relB + (size_t)relIdx[e0] * 256u;
    const unsigned char* pT0 = timB + (size_t)timIdx[e0] * 256u;
    const unsigned char* pH1 = hB   + (size_t)srcIdx[e1] * 256u;
    const unsigned char* pR1 = relB + (size_t)relIdx[e1] * 256u;
    const unsigned char* pT1 = timB + (size_t)timIdx[e1] * 256u;

    v8f C[2][8];
#pragma unroll
    for (int a = 0; a < 2; ++a)
#pragma unroll
      for (int t = 0; t < 8; ++t) C[a][t] = zero8;

#pragma unroll
    for (int seg = 0; seg < 3; ++seg) {      // h_s | h_r | h_t (compile-time select)
      const unsigned char* q0 = (seg == 0) ? pH0 : (seg == 1) ? pR0 : pT0;
      const unsigned char* q1 = (seg == 0) ? pH1 : (seg == 1) ? pR1 : pT1;
#pragma unroll 1
      for (int ss = 0; ss < 4; ++ss) {       // 4 k-steps of 32 within the segment
        const int off0 = ss * 64 + half * 32;     // contiguous k = 16*half + j
        BfPack A0, A1;
        A0.q[0] = *(gc_v4u*)(q0 + off0);
        A0.q[1] = *(gc_v4u*)(q0 + off0 + 16);
        A1.q[0] = *(gc_v4u*)(q1 + off0);
        A1.q[1] = *(gc_v4u*)(q1 + off0 + 16);
#pragma unroll
        for (int t = 0; t < 8; ++t) {        // all 8 n-tiles, one pass over K
          const int s = seg * 4 + ss;
          BfPack Bv;
          const v4u* lp = ldsW + ((t * 12 + s) * 64 + lane * 2);
          Bv.q[0] = lp[0];
          Bv.q[1] = lp[1];
          C[0][t] = __builtin_amdgcn_wmma_f32_16x16x32_bf16(
              false, A0.v, false, Bv.v, (short)0, C[0][t], false, false);
          C[1][t] = __builtin_amdgcn_wmma_f32_16x16x32_bf16(
              false, A1.v, false, Bv.v, (short)0, C[1][t], false, false);
        }
      }
    }

    // epilogue: alpha[m] = sum_n relu(C + cvec[n]) * W2[n]; butterfly; sigmoid
#pragma unroll
    for (int a = 0; a < 2; ++a) {
      float acc[8];
#pragma unroll
      for (int v = 0; v < 8; ++v) acc[v] = 0.0f;
#pragma unroll
      for (int t = 0; t < 8; ++t)
#pragma unroll
        for (int v = 0; v < 8; ++v) {
          float hval = C[a][t][v] + cv[t];
          hval = fmaxf(hval, 0.0f);
          acc[v] += hval * w2v[t];
        }
#pragma unroll
      for (int m = 1; m <= 8; m <<= 1) {
        acc[0] += __shfl_xor(acc[0], m, 32);
        acc[1] += __shfl_xor(acc[1], m, 32);
        acc[2] += __shfl_xor(acc[2], m, 32);
        acc[3] += __shfl_xor(acc[3], m, 32);
        acc[4] += __shfl_xor(acc[4], m, 32);
        acc[5] += __shfl_xor(acc[5], m, 32);
        acc[6] += __shfl_xor(acc[6], m, 32);
        acc[7] += __shfl_xor(acc[7], m, 32);
      }
      const int r = lane & 7;
      float val = acc[0];
      if (r == 1) val = acc[1];
      if (r == 2) val = acc[2];
      if (r == 3) val = acc[3];
      if (r == 4) val = acc[4];
      if (r == 5) val = acc[5];
      if (r == 6) val = acc[6];
      if (r == 7) val = acc[7];
      if ((lane & 8) == 0) {                  // lanes 0-7 -> rows 0-7, 16-23 -> rows 8-15
        const int m = half * 8 + r;
        const int e = ebase + a * 16 + m;
        if (e < E) {
          float z = val + bias2;
          out[e] = 1.0f / (1.0f + __expf(-z));
        }
      }
    }
  }
}

// ---------- launch ----------
extern "C" void kernel_launch(void* const* d_in, const int* in_sizes, int n_in,
                              void* d_out, int out_size, void* d_ws, size_t ws_size,
                              hipStream_t stream) {
  const float* h    = (const float*)d_in[0];
  const int*   eidx = (const int*)d_in[1];    // [2,E]; row 0 = src
  const int*   etyp = (const int*)d_in[2];
  const int*   etim = (const int*)d_in[3];
  const int*   qrel = (const int*)d_in[4];
  const float* rel  = (const float*)d_in[5];
  const float* tim  = (const float*)d_in[6];
  const float* W1   = (const float*)d_in[7];
  const float* b1   = (const float*)d_in[8];
  const float* W2   = (const float*)d_in[9];
  const float* b2   = (const float*)d_in[10];
  float* out = (float*)d_out;

  const int E = in_sizes[2];
  const int N = in_sizes[0] / 128;
  const int R = in_sizes[5] / 128;
  const int T = in_sizes[6] / 128;

  unsigned char* ws = (unsigned char*)d_ws;
  size_t off = 0;
  auto carve = [&](size_t bytes) -> unsigned char* {
    off = (off + 255) & ~(size_t)255;
    unsigned char* p = ws + off;
    off += bytes;
    return p;
  };
  unsigned short* hB   = (unsigned short*)carve((size_t)N * 256);
  unsigned short* relB = (unsigned short*)carve((size_t)R * 256);
  unsigned short* timB = (unsigned short*)carve((size_t)T * 256);
  unsigned short* W1p  = (unsigned short*)carve(98304);
  float*          cvec = (float*)carve(512);

  { int n8 = N * 16; cvt_bf16_kernel<<<(n8 + 255) / 256, 256, 0, stream>>>(h,   (uint4*)hB,   n8); }
  { int n8 = R * 16; cvt_bf16_kernel<<<(n8 + 255) / 256, 256, 0, stream>>>(rel, (uint4*)relB, n8); }
  { int n8 = T * 16; cvt_bf16_kernel<<<(n8 + 255) / 256, 256, 0, stream>>>(tim, (uint4*)timB, n8); }
  pack_w1_kernel<<<192, 256, 0, stream>>>(W1, W1p, 49152);
  calc_cvec_kernel<<<1, 128, 0, stream>>>(rel, W1, b1, qrel, cvec);

  const int tiles = (E + 31) / 32;
  int blocks = 1024;
  if (blocks * 8 > tiles) blocks = (tiles + 7) / 8;
  if (blocks < 1) blocks = 1;
  edge_score_kernel<<<blocks, 256, 98304, stream>>>(
      out, eidx, etyp, etim,
      (const unsigned char*)hB, (const unsigned char*)relB, (const unsigned char*)timB,
      (const v4u*)W1p, cvec, W2, b2, E);
}
